// GATEncoder_49993419325828
// MI455X (gfx1250) — compile-verified, matrix operand order checked by reading, not run on hardware
//
#include <hip/hip_runtime.h>

// ---------------------------------------------------------------------------
// GAT encoder for MI455X (gfx1250):
//   - all GEMMs via v_wmma_f32_16x16x32_bf16 (f32 accumulate)
//   - GEMM tiles streamed global->LDS with the Tensor Data Mover
//     (tensor_load_to_lds + s_wait_tensorcnt), double-buffered
//   - B matrices stored transposed so both fragments are contiguous b128 LDS reads
// ---------------------------------------------------------------------------

typedef __attribute__((ext_vector_type(16))) __bf16 v16bf;
typedef __attribute__((ext_vector_type(8)))  float  v8f;
typedef __attribute__((ext_vector_type(4)))  unsigned int v4u;
typedef __attribute__((ext_vector_type(8)))  int v8i;
typedef __attribute__((ext_vector_type(4)))  int v4i;

#if __has_builtin(__builtin_amdgcn_tensor_load_to_lds) && \
    __has_builtin(__builtin_amdgcn_s_wait_tensorcnt)
#define HAVE_TDM 1
#else
#define HAVE_TDM 0
#endif

#define GEMM_BM 128
#define GEMM_BN 64
#define GEMM_BK 32

// Problem constants (from reference setup_inputs)
#define NN   2048   // nodes
#define FF   300    // input features
#define HH   20     // heads
#define HID  600    // per-head dim
#define HIDP 640    // padded per-head dim (x64)
#define OUTD 1000   // out_att dim
#define OUTP 1024   // padded (x64)
#define MLP  1000
#define MLPP 1024
#define NOUT 768
#define NG   128    // graphs
#define KX   320    // padded F (x32)
#define KH   (HH*HIDP)  // 12800, padded concat dim

// ---------------------------------------------------------------------------
// Generic bf16 WMMA GEMM: C[M,N](f32) = A[M,K](bf16 row-major) x BT[N,K](bf16)
// (B supplied TRANSPOSED: BT[n][k], leading dim ldBT = K-stride).
// Requires: M % 128 == 0, N % 64 == 0, K % 32 == 0 (buffers are padded).
// act: 0 = none, 1 = relu, 2 = elu.  bias applied for col < Nb.
// ---------------------------------------------------------------------------
__global__ __launch_bounds__(256) void gemm_bf16_wmma(
    const __bf16* __restrict__ A, const __bf16* __restrict__ BT,
    float* __restrict__ C, int M, int N, int K,
    int ldA, int ldBT, int ldC,
    const float* __restrict__ bias, int Nb, int act)
{
  __shared__ __bf16 sA[2][GEMM_BM][GEMM_BK];   // 2 x 8 KB
  __shared__ __bf16 sB[2][GEMM_BN][GEMM_BK];   // 2 x 4 KB
  const int tid  = threadIdx.x;
  const int wave = tid >> 5;        // 8 waves: wave w owns rows [w*16, w*16+16)
  const int lane = tid & 31;
  const int l16  = lane & 15;
  const int koff = (lane >> 4) * 8; // ISA 16-bit frag layout: half-wave K offset
  const int bm = blockIdx.y * GEMM_BM;
  const int bn = blockIdx.x * GEMM_BN;

  v8f acc[4];
#pragma unroll
  for (int t = 0; t < 4; ++t)
#pragma unroll
    for (int r = 0; r < 8; ++r) acc[t][r] = 0.0f;

  // 16x16x32 step on double-buffered LDS tiles
  auto mma_step = [&](int buf) {
    v16bf afrag;
    {
      const __bf16* ap = &sA[buf][wave * 16 + l16][0];
#pragma unroll
      for (int i = 0; i < 8; ++i) {
        afrag[i]     = ap[koff + i];
        afrag[8 + i] = ap[16 + koff + i];
      }
    }
#pragma unroll
    for (int nt = 0; nt < 4; ++nt) {
      v16bf bfrag;
      const __bf16* bp = &sB[buf][nt * 16 + l16][0];
#pragma unroll
      for (int i = 0; i < 8; ++i) {
        bfrag[i]     = bp[koff + i];
        bfrag[8 + i] = bp[16 + koff + i];
      }
      acc[nt] = __builtin_amdgcn_wmma_f32_16x16x32_bf16(
          false, afrag, false, bfrag, (short)0, acc[nt], false, false);
    }
  };

#if HAVE_TDM
  // ---- Tensor Data Mover descriptor issue (D# per CDNA5 ISA) ----
  // 2D tile: GEMM_BK contiguous bf16 per row, `rows` rows, row stride ldg.
  auto issue_tile = [&](const __bf16* gptr, unsigned ldsoff, int rows, int ldg) {
    unsigned long long ga = (unsigned long long)(const void*)gptr;
    v4u g0;
    g0[0] = 1u;                                        // count=1 (valid D#)
    g0[1] = ldsoff;                                    // lds_addr (bytes)
    g0[2] = (unsigned)(ga & 0xFFFFFFFFull);            // global_addr[31:0]
    g0[3] = (unsigned)((ga >> 32) & 0x01FFFFFFull)     // global_addr[56:32]
            | 0x80000000u;                             // type=2 ("image")
    v8i g1;
    g1[0] = 0x00010000;                                // data_size=1 (2 bytes)
    g1[1] = (int)0xFFFF0000u;                          // tensor_dim0 = 0x7FFFFFFF (lo16)
    g1[2] = (int)0xFFFF7FFFu;                          // td0 hi16 | tensor_dim1 lo16
    g1[3] = (int)(0x7FFFu | ((unsigned)GEMM_BK << 16));// td1 hi16 | tile_dim0=32
    g1[4] = rows;                                      // tile_dim1 (tile_dim2=0)
    g1[5] = ldg;                                       // tensor_dim0_stride[31:0]
    g1[6] = 0;                                         // stride0 hi | stride1 lo
    g1[7] = 0;                                         // stride1 hi
    v4i gz4; gz4[0] = 0; gz4[1] = 0; gz4[2] = 0; gz4[3] = 0;
    v8i gz8;
#pragma unroll
    for (int i = 0; i < 8; ++i) gz8[i] = 0;
    // 6-arg form (clang-23 / therock-10.0 headers)
    __builtin_amdgcn_tensor_load_to_lds(g0, g1, gz4, gz4, gz8, 0);
  };
  const bool issuer = (wave == 0);     // one wave issues; TENSORcnt is per-wave
  const unsigned sA0 = (unsigned)(size_t)&sA[0][0][0]; // generic LDS addr low 32
  const unsigned sA1 = (unsigned)(size_t)&sA[1][0][0]; //   bits == LDS offset
  const unsigned sB0 = (unsigned)(size_t)&sB[0][0][0];
  const unsigned sB1 = (unsigned)(size_t)&sB[1][0][0];

  const int nk = K / GEMM_BK;
  if (issuer) {
    issue_tile(A  + (size_t)bm * ldA,  sA0, GEMM_BM, ldA);
    issue_tile(BT + (size_t)bn * ldBT, sB0, GEMM_BN, ldBT);
  }
  for (int t = 0; t < nk; ++t) {
    const int buf = t & 1;
    if (t + 1 < nk) {
      if (issuer) {
        const int k0n = (t + 1) * GEMM_BK;
        issue_tile(A  + (size_t)bm * ldA  + k0n, buf ? sA0 : sA1, GEMM_BM, ldA);
        issue_tile(BT + (size_t)bn * ldBT + k0n, buf ? sB0 : sB1, GEMM_BN, ldBT);
      }
      __builtin_amdgcn_s_wait_tensorcnt(2);  // pair(t) done, pair(t+1) in flight
    } else {
      __builtin_amdgcn_s_wait_tensorcnt(0);
    }
    __syncthreads();      // tile t visible to all waves
    mma_step(buf);
    __syncthreads();      // all waves done with buf before it is re-filled
  }
#else
  // ---- fallback: synchronous single-buffer staging ----
  for (int k0 = 0; k0 < K; k0 += GEMM_BK) {
    __syncthreads();
    for (int q = tid; q < (GEMM_BM * GEMM_BK) / 8; q += 256) {
      int row = q >> 2, ch = q & 3;
      *(uint4*)(&sA[0][row][ch * 8]) =
          *(const uint4*)(A + (size_t)(bm + row) * ldA + k0 + ch * 8);
    }
    {
      int row = tid >> 2, ch = tid & 3;   // 64 rows x 4 chunks of 8 bf16
      *(uint4*)(&sB[0][row][ch * 8]) =
          *(const uint4*)(BT + (size_t)(bn + row) * ldBT + k0 + ch * 8);
    }
    __syncthreads();
    mma_step(0);
  }
#endif

  // C layout: VGPR r -> row r (lanes 0-15) / row r+8 (lanes 16-31), col = lane%16
  const int rowbase = bm + wave * 16 + (lane >> 4) * 8;
#pragma unroll
  for (int nt = 0; nt < 4; ++nt) {
    int col = bn + nt * 16 + l16;
    float bv = (bias != nullptr && col < Nb) ? bias[col] : 0.0f;
#pragma unroll
    for (int r = 0; r < 8; ++r) {
      float v = acc[nt][r] + bv;
      if (act == 1)      v = fmaxf(v, 0.0f);
      else if (act == 2) v = (v > 0.0f) ? v : expm1f(v);
      C[(size_t)(rowbase + r) * ldC + col] = v;
    }
  }
}

// ---------------------------------------------------------------------------
// f32 -> bf16 convert with zero padding. dst[rt x ct], src[rv x cv].
// ---------------------------------------------------------------------------
__global__ void cvt_pad(const float* __restrict__ src, int lds, int rv, int cv,
                        __bf16* __restrict__ dst, int ldd, int rt, int ct)
{
  long id = (long)blockIdx.x * blockDim.x + threadIdx.x;
  if (id >= (long)rt * ct) return;
  int r = (int)(id / ct);
  int c = (int)(id % ct);
  float v = (r < rv && c < cv) ? src[(size_t)r * lds + c] : 0.0f;
  dst[(size_t)r * ldd + c] = (__bf16)v;
}

// f32 [K x N] -> bf16 TRANSPOSED dst[NT x KT] (dst[n*KT+k] = src[k*lds+n]).
__global__ void cvt_padT(const float* __restrict__ src, int lds, int rv, int cv,
                         __bf16* __restrict__ dst, int KT, long total)
{
  long id = (long)blockIdx.x * blockDim.x + threadIdx.x;
  if (id >= total) return;
  int n = (int)(id / KT);
  int k = (int)(id % KT);
  float v = (k < rv && n < cv) ? src[(size_t)k * lds + n] : 0.0f;
  dst[id] = (__bf16)v;
}

// W_out remap + transpose: dst[OUTP][KH]; k = head*640 + c0 -> src row head*600+c0.
__global__ void cvt_woutT(const float* __restrict__ W, __bf16* __restrict__ dst)
{
  long id = (long)blockIdx.x * blockDim.x + threadIdx.x;
  if (id >= (long)OUTP * KH) return;
  int n = (int)(id / KH);
  int r = (int)(id % KH);
  int head = r / HIDP, c0 = r % HIDP;
  float v = (c0 < HID && n < OUTD) ? W[(size_t)(head * HID + c0) * OUTD + n] : 0.0f;
  dst[id] = (__bf16)v;
}

// adjacency bitmask scatter: adj[src*64 + dst/32] |= 1<<(dst%32)
__global__ void edge_scatter(const int* __restrict__ ei, int E,
                             unsigned* __restrict__ adj)
{
  int e = blockIdx.x * blockDim.x + threadIdx.x;
  if (e >= E) return;
  int s = ei[e];
  int d = ei[E + e];
  atomicOr(&adj[(size_t)s * (NN >> 5) + (d >> 5)], 1u << (d & 31));
}

// f1[i] = Wh[i,:hid] . a1 ; f2[i] = Wh[i,:hid] . a2
__global__ void fvec(const float* __restrict__ Wh, int ld, int hid,
                     const float* __restrict__ a1, const float* __restrict__ a2,
                     float* __restrict__ f1, float* __restrict__ f2)
{
  int i = blockIdx.x * blockDim.x + threadIdx.x;
  if (i >= NN) return;
  const float* w = Wh + (size_t)i * ld;
  float s1 = 0.0f, s2 = 0.0f;
  for (int k = 0; k < hid; ++k) {
    float v = w[k];
    s1 += v * a1[k];
    s2 += v * a2[k];
  }
  f1[i] = s1;
  f2[i] = s2;
}

// Row-wise masked leaky-relu softmax -> bf16 attention matrix. Block per row.
__global__ __launch_bounds__(256) void att_softmax(
    const float* __restrict__ f1, const float* __restrict__ f2,
    const unsigned* __restrict__ adj, __bf16* __restrict__ att)
{
  const int row = blockIdx.x;
  const int tid = threadIdx.x;
  __shared__ float red[256];
  const float f1r = f1[row];
  const unsigned* arow = adj + (size_t)row * (NN >> 5);
  float vals[NN / 256];

  float m = -3.0e38f;
#pragma unroll
  for (int i = 0; i < NN / 256; ++i) {
    int j = tid + (i << 8);
    unsigned bit = (arow[j >> 5] >> (j & 31)) & 1u;
    float e = f1r + f2[j];
    e = (e >= 0.0f) ? e : 0.02f * e;          // leaky_relu, slope 0.02
    float v = bit ? e : -9.0e15f;             // reference NEG_INF
    vals[i] = v;
    m = fmaxf(m, v);
  }
  red[tid] = m; __syncthreads();
  for (int s = 128; s > 0; s >>= 1) {
    if (tid < s) red[tid] = fmaxf(red[tid], red[tid + s]);
    __syncthreads();
  }
  m = red[0];
  __syncthreads();

  float ssum = 0.0f;
#pragma unroll
  for (int i = 0; i < NN / 256; ++i) {
    vals[i] = expf(vals[i] - m);
    ssum += vals[i];
  }
  red[tid] = ssum; __syncthreads();
  for (int s = 128; s > 0; s >>= 1) {
    if (tid < s) red[tid] += red[tid + s];
    __syncthreads();
  }
  const float inv = 1.0f / red[0];
#pragma unroll
  for (int i = 0; i < NN / 256; ++i)
    att[(size_t)row * NN + tid + (i << 8)] = (__bf16)(vals[i] * inv);
}

// Pooling
__global__ void pool_count(const int* __restrict__ batch, float* __restrict__ counts)
{
  int i = blockIdx.x * blockDim.x + threadIdx.x;
  if (i < NN) atomicAdd(&counts[batch[i]], 1.0f);
}

__global__ void pool_acc(const float* __restrict__ outf, const int* __restrict__ batch,
                         float* __restrict__ sums)
{
  long id = (long)blockIdx.x * blockDim.x + threadIdx.x;
  if (id >= (long)NN * OUTD) return;
  int r = (int)(id / OUTD);
  int c = (int)(id % OUTD);
  float v = outf[(size_t)r * OUTP + c];       // elu already fused into GEMM
  atomicAdd(&sums[(size_t)batch[r] * OUTP + c], v);
}

__global__ void pool_fin(const float* __restrict__ sums, const float* __restrict__ counts,
                         __bf16* __restrict__ pooled)
{
  int id = blockIdx.x * blockDim.x + threadIdx.x;
  if (id >= NG * OUTP) return;
  int g = id >> 10, c = id & 1023;
  float v = (c < OUTD) ? sums[id] / fmaxf(counts[g], 1.0f) : 0.0f;
  pooled[id] = (__bf16)v;
}

// ---------------------------------------------------------------------------

static inline void launch_gemm(const __bf16* A, const __bf16* BT, float* C,
                               int M, int N, int K, int ldA, int ldBT, int ldC,
                               const float* bias, int Nb, int act,
                               hipStream_t stream)
{
  dim3 grid(N / GEMM_BN, M / GEMM_BM);
  gemm_bf16_wmma<<<grid, 256, 0, stream>>>(A, BT, C, M, N, K, ldA, ldBT, ldC,
                                           bias, Nb, act);
}

static inline void launch_cvt(const float* src, int lds, int rv, int cv,
                              __bf16* dst, int ldd, int rt, int ct,
                              hipStream_t stream)
{
  long total = (long)rt * ct;
  cvt_pad<<<(int)((total + 255) / 256), 256, 0, stream>>>(src, lds, rv, cv,
                                                          dst, ldd, rt, ct);
}

static inline void launch_cvtT(const float* src, int lds, int rv, int cv,
                               __bf16* dst, int KT, int NT, hipStream_t stream)
{
  long total = (long)NT * KT;
  cvt_padT<<<(int)((total + 255) / 256), 256, 0, stream>>>(src, lds, rv, cv,
                                                           dst, KT, total);
}

extern "C" void kernel_launch(void* const* d_in, const int* in_sizes, int n_in,
                              void* d_out, int out_size, void* d_ws, size_t ws_size,
                              hipStream_t stream)
{
  const float* x      = (const float*)d_in[0];
  const int*   edge   = (const int*)d_in[1];
  const int*   batch  = (const int*)d_in[2];
  const float* Wheads = (const float*)d_in[3];
  const float* a1h    = (const float*)d_in[4];
  const float* a2h    = (const float*)d_in[5];
  const float* Wout   = (const float*)d_in[6];
  const float* a1o    = (const float*)d_in[7];
  const float* a2o    = (const float*)d_in[8];
  const float* W1     = (const float*)d_in[9];
  const float* b1     = (const float*)d_in[10];
  const float* W2     = (const float*)d_in[11];
  const float* b2     = (const float*)d_in[12];
  float* out = (float*)d_out;
  const int E = in_sizes[1] / 2;

  // ---- workspace carve (aligned 256B) ----
  char* p = (char*)d_ws;
  auto carve = [&](size_t bytes) -> void* {
    void* r = (void*)p;
    p += (bytes + 255) & ~(size_t)255;
    return r;
  };
  unsigned* adjbits = (unsigned*)carve((size_t)NN * (NN / 32) * 4);   // 512 KB
  __bf16* xbf    = (__bf16*)carve((size_t)NN * KX * 2);               // x padded
  __bf16* WbfT   = (__bf16*)carve((size_t)HIDP * KX * 2);             // per-head W^T
  float*  Whf    = (float*)carve((size_t)NN * OUTP * 4);              // f32 scratch
  __bf16* WhbfT  = (__bf16*)carve((size_t)OUTP * NN * 2);             // Wh^T bf16
  float*  f1     = (float*)carve(NN * 4);
  float*  f2     = (float*)carve(NN * 4);
  __bf16* attbf  = (__bf16*)carve((size_t)NN * NN * 2);               // 8 MB
  __bf16* hbf    = (__bf16*)carve((size_t)NN * KH * 2);               // concat heads
  __bf16* woutbfT = (__bf16*)carve((size_t)OUTP * KH * 2);
  float*  sums   = (float*)carve((size_t)NG * OUTP * 4);
  float*  counts = (float*)carve(NG * 4);
  __bf16* pooledbf = (__bf16*)carve((size_t)NG * OUTP * 2);
  __bf16* w1bfT  = (__bf16*)carve((size_t)MLPP * MLPP * 2);
  float*  hidf   = (float*)carve((size_t)NG * MLPP * 4);
  __bf16* hidbf  = (__bf16*)carve((size_t)NG * MLPP * 2);
  __bf16* w2bfT  = (__bf16*)carve((size_t)NOUT * MLPP * 2);
  (void)ws_size; (void)n_in; (void)out_size;

  // ---- adjacency bitmask ----
  (void)hipMemsetAsync(adjbits, 0, (size_t)NN * (NN / 32) * 4, stream);
  edge_scatter<<<(E + 255) / 256, 256, 0, stream>>>(edge, E, adjbits);

  // ---- one-time conversions (A row-major, B transposed) ----
  launch_cvt(x, FF, NN, FF, xbf, KX, NN, KX, stream);
  launch_cvtT(W1, MLP, MLP, MLP, w1bfT, MLPP, MLPP, stream);       // [1024][1024]
  launch_cvtT(W2, NOUT, MLP, NOUT, w2bfT, MLPP, NOUT, stream);     // [768][1024]
  {
    long total = (long)OUTP * KH;
    cvt_woutT<<<(int)((total + 255) / 256), 256, 0, stream>>>(Wout, woutbfT);
  }

  // ---- per-head GAT layer ----
  for (int h = 0; h < HH; ++h) {
    // W_h [300,600] -> WbfT [640][320]
    launch_cvtT(Wheads + (size_t)h * FF * HID, HID, FF, HID, WbfT, KX, HIDP, stream);
    // Wh = x @ W_h : [2048,320] x (320,640)
    launch_gemm(xbf, WbfT, Whf, NN, HIDP, KX, KX, KX, HIDP,
                nullptr, 0, 0, stream);
    fvec<<<NN / 256, 256, 0, stream>>>(Whf, HIDP, HID,
                                       a1h + (size_t)h * HID,
                                       a2h + (size_t)h * HID, f1, f2);
    // Wh [2048,640] -> Wh^T bf16 [640][2048]
    launch_cvtT(Whf, HIDP, NN, HIDP, WhbfT, NN, HIDP, stream);
    att_softmax<<<NN, 256, 0, stream>>>(f1, f2, adjbits, attbf);
    // head_out = elu(att @ Wh) : [2048,2048] x (2048,640), ELU fused
    launch_gemm(attbf, WhbfT, Whf, NN, HIDP, NN, NN, NN, HIDP,
                nullptr, 0, 2, stream);
    // concat into h (bf16), zero the 600..639 pad
    launch_cvt(Whf, HIDP, NN, HID, hbf + (size_t)h * HIDP, KH, NN, HIDP, stream);
  }

  // ---- output attention layer ----
  // Wh_out = h @ W_out : [2048,12800] x (12800,1024)
  launch_gemm(hbf, woutbfT, Whf, NN, OUTP, KH, KH, KH, OUTP,
              nullptr, 0, 0, stream);
  fvec<<<NN / 256, 256, 0, stream>>>(Whf, OUTP, OUTD, a1o, a2o, f1, f2);
  launch_cvtT(Whf, OUTP, NN, OUTP, WhbfT, NN, OUTP, stream);       // [1024][2048]
  att_softmax<<<NN, 256, 0, stream>>>(f1, f2, adjbits, attbf);
  // out = elu(att @ Wh_out), ELU fused
  launch_gemm(attbf, WhbfT, Whf, NN, OUTP, NN, NN, NN, OUTP,
              nullptr, 0, 2, stream);

  // ---- global mean pool ----
  (void)hipMemsetAsync(sums, 0, (size_t)NG * OUTP * 4, stream);
  (void)hipMemsetAsync(counts, 0, NG * 4, stream);
  pool_count<<<NN / 256, 256, 0, stream>>>(batch, counts);
  {
    long total = (long)NN * OUTD;
    pool_acc<<<(int)((total + 255) / 256), 256, 0, stream>>>(Whf, batch, sums);
  }
  pool_fin<<<(NG * OUTP) / 256, 256, 0, stream>>>(sums, counts, pooledbf);

  // ---- MLP head ----
  // hid = relu(pooled @ W1 + b1) : [128,1024] x (1024,1024)
  launch_gemm(pooledbf, w1bfT, hidf, NG, MLPP, OUTP, OUTP, MLPP, MLPP,
              b1, MLP, 1, stream);
  launch_cvt(hidf, MLPP, NG, MLPP, hidbf, MLPP, NG, MLPP, stream);
  // out = hid @ W2 + b2 -> d_out directly : [128,1024] x (1024,768)
  launch_gemm(hidbf, w2bfT, out, NG, NOUT, MLPP, MLPP, MLPP, NOUT,
              b2, NOUT, 0, stream);
}
